// LocalWindowAttention_55585466745411
// MI455X (gfx1250) — matrix operand order repaired and checked by reading, not compiled, not run on hardware
//
#include <hip/hip_runtime.h>
#include <hip/hip_bf16.h>
#include <stdint.h>

// ---------------------------------------------------------------------------
// Fused channel-reduced spatial attention for MI455X (gfx1250, wave32, WMMA).
// B=8, C=256, H=W=64 -> N=4096, r=32.
//
//   q = Wq x + bq   (B,N,32)  bf16, token-major
//   k = Wk x + bk   (B,N,32)  bf16, token-major
//   v = Wv x + bv   (B,C,N)   bf16, C-major (rows contiguous along tokens)
//   out = softmax(q k^T / sqrt(r)) v^T ; y = x + gamma*out
//
// Flash-attention style: the 4096x4096 P matrix is never written to HBM.
// Matmuls on v_wmma_f32_16x16x32_bf16. K/V tiles are DMA'd into LDS by the
// Tensor Data Mover, DOUBLE-BUFFERED so tile mt+1 streams in while tile mt
// is being consumed (TENSORcnt + workgroup barrier handshake).
// ---------------------------------------------------------------------------

typedef __attribute__((ext_vector_type(16))) __bf16 v16bf;
typedef __attribute__((ext_vector_type(8)))  __bf16 v8bf;
typedef __attribute__((ext_vector_type(8)))  float  v8f;
typedef __attribute__((ext_vector_type(4)))  unsigned int v4u;
typedef __attribute__((ext_vector_type(8)))  int    v8i;
typedef __attribute__((ext_vector_type(4)))  int    v4i;

#define BATCH 8
#define CCH   256
#define NTOK  4096
#define RDIM  32

__device__ __forceinline__ v16bf frag16(const __bf16* p0, const __bf16* p1) {
  v8bf lo = *(const v8bf*)p0;
  v8bf hi = *(const v8bf*)p1;
  v16bf r;
#pragma unroll
  for (int i = 0; i < 8; ++i) { r[i] = lo[i]; r[i + 8] = hi[i]; }
  return r;
}

__device__ __forceinline__ v8f wmma_bf16(v16bf a, v16bf b, v8f c) {
  return __builtin_amdgcn_wmma_f32_16x16x32_bf16(
      false, a, false, b, (short)0, c, false, false);
}

#if __has_builtin(__builtin_amdgcn_tensor_load_to_lds) && \
    __has_builtin(__builtin_amdgcn_s_wait_tensorcnt)
#define USE_TDM 1
#else
#define USE_TDM 0
#endif

#if USE_TDM
// 2D TDM load: tile (tile_d0 contiguous elems) x (tile_d1 rows) of 2-byte
// elements from a row-major tensor with row stride stride_d0 (elements) into
// LDS at lds_byte, padding each row by pad_amount dwords every 2^(pad_int+1)
// dwords (D# per CDNA5 ISA). 6-arg builtin form:
// (v4u g0, v8i g1, v4i g2, v4i g3, v8i g4, i32 cpol).
__device__ __forceinline__ void tdm_load_2d_bf16(
    uint32_t lds_byte, const void* gptr,
    uint32_t tile_d0, uint32_t tile_d1,
    uint32_t tensor_d0, uint32_t tensor_d1, uint64_t stride_d0,
    uint32_t pad_interval_code, uint32_t pad_amount_code) {
  uint64_t ga = (uint64_t)(uintptr_t)gptr;
  v4u g0;
  g0[0] = 1u;                                   // count=1, user mode
  g0[1] = lds_byte;                             // lds_addr
  g0[2] = (uint32_t)ga;                         // global_addr[31:0]
  g0[3] = (uint32_t)((ga >> 32) & 0x01ffffffu)  // global_addr[56:32]
          | (2u << 30);                         // type=2 (image)
  v8i g1;
  g1[0] = (int)((1u << 16)                      // data_size: 2 bytes
                | (1u << 20)                    // pad_enable
                | (pad_interval_code << 22)
                | (pad_amount_code << 25));     // workgroup_mask=0
  g1[1] = (int)((tensor_d0 & 0xffffu) << 16);   // ab_addr=0 | tensor_dim0 lo
  g1[2] = (int)(((tensor_d0 >> 16) & 0xffffu) | ((tensor_d1 & 0xffffu) << 16));
  g1[3] = (int)(((tensor_d1 >> 16) & 0xffffu) | ((tile_d0 & 0xffffu) << 16));
  g1[4] = (int)(tile_d1 & 0xffffu);             // tile_dim1 | tile_dim2=0
  g1[5] = (int)(uint32_t)(stride_d0 & 0xffffffffu);   // dim0_stride[31:0]
  g1[6] = (int)(uint32_t)((stride_d0 >> 32) & 0xffffu); // dim0_stride[47:32]
  g1[7] = 0;                                    // tensor_dim1_stride (2D: n/a)
  v4i gz = {};
  v8i gz8 = {};
  __builtin_amdgcn_tensor_load_to_lds(g0, g1, gz, gz, gz8, 0);
}
#endif

// ---------------------------------------------------------------------------
// Kernel 0: one-time f32 -> bf16 conversion of the projection weights.
// ---------------------------------------------------------------------------
__global__ void convert_weights(const float* __restrict__ wq,
                                const float* __restrict__ wk,
                                const float* __restrict__ wv,
                                __bf16* __restrict__ wq_b,
                                __bf16* __restrict__ wk_b,
                                __bf16* __restrict__ wv_b) {
  int i = blockIdx.x * blockDim.x + threadIdx.x;
  if (i < RDIM * CCH) {
    wq_b[i] = (__bf16)wq[i];
    wk_b[i] = (__bf16)wk[i];
  }
  if (i < CCH * CCH) {
    wv_b[i] = (__bf16)wv[i];
  }
}

// ---------------------------------------------------------------------------
// Kernel 1: projections. One workgroup = one (batch, 64-token tile).
// 8 waves: waves 0-3 compute q + v[:,0:128], waves 4-7 compute k + v[:,128:256].
// ---------------------------------------------------------------------------
#define XS_STRIDE 264   // 64 tokens x 256 ch, padded; 264*2B is a multiple of 16

__global__ __launch_bounds__(256)
void proj_kernel(const float* __restrict__ x,
                 const __bf16* __restrict__ wq_b,
                 const __bf16* __restrict__ wk_b,
                 const __bf16* __restrict__ wv_b,
                 const float* __restrict__ bq,
                 const float* __restrict__ bk,
                 const float* __restrict__ bv,
                 __bf16* __restrict__ qg,
                 __bf16* __restrict__ kg,
                 __bf16* __restrict__ vg) {
  __shared__ __align__(16) __bf16 xs[64 * XS_STRIDE];

  const int b     = blockIdx.x >> 6;
  const int nbase = (blockIdx.x & 63) * 64;
  const int tid   = threadIdx.x;

  // Stage x tile -> LDS, token-major bf16 (transpose from C-major global).
#pragma unroll 4
  for (int it = 0; it < 64; ++it) {
    int idx = it * 256 + tid;
    int n = idx & 63;
    int c = idx >> 6;
    xs[n * XS_STRIDE + c] = (__bf16)x[(b * CCH + c) * NTOK + nbase + n];
  }
  __syncthreads();

  const int w    = tid >> 5;
  const int lane = tid & 31;
  const int l16  = lane & 15;
  const int half = lane >> 4;
  const int rt   = w & 3;           // 16-row tile within the 64-token tile
  const int rowb = rt * 16;

  const __bf16* Wpk   = (w < 4) ? wq_b : wk_b;   // q or k weights
  const int     vcol0 = (w < 4) ? 0 : 128;       // v column half

  v8f accP[2]; v8f accV[8];
#pragma unroll
  for (int t = 0; t < 2; ++t) accP[t] = {};
#pragma unroll
  for (int t = 0; t < 8; ++t) accV[t] = {};

  const int arow = (rowb + l16) * XS_STRIDE;
  for (int kc = 0; kc < CCH; kc += 32) {
    v16bf a = frag16(&xs[arow + kc + half * 16],
                     &xs[arow + kc + half * 16 + 8]);
    v16bf bp[2], bvf[8];
#pragma unroll
    for (int t = 0; t < 2; ++t) {
      const __bf16* wp = &Wpk[(t * 16 + l16) * CCH + kc + half * 16];
      bp[t] = frag16(wp, wp + 8);
    }
#pragma unroll
    for (int t = 0; t < 8; ++t) {
      const __bf16* wp = &wv_b[(vcol0 + t * 16 + l16) * CCH + kc + half * 16];
      bvf[t] = frag16(wp, wp + 8);
    }
#pragma unroll
    for (int t = 0; t < 2; ++t) accP[t] = wmma_bf16(a, bp[t], accP[t]);
#pragma unroll
    for (int t = 0; t < 8; ++t) accV[t] = wmma_bf16(a, bvf[t], accV[t]);
  }

  // Store q/k (token-major) and v (C-major), with bias.
  const float*  biasP = (w < 4) ? bq : bk;
  __bf16*       outP  = (w < 4) ? qg : kg;
#pragma unroll
  for (int t = 0; t < 2; ++t) {
    int col = t * 16 + l16;
    float bias = biasP[col];
#pragma unroll
    for (int j = 0; j < 8; ++j) {
      int n = nbase + rowb + j + 8 * half;   // D layout: VGPR j -> rows j / j+8
      outP[(b * NTOK + n) * RDIM + col] = (__bf16)(accP[t][j] + bias);
    }
  }
#pragma unroll
  for (int t = 0; t < 8; ++t) {
    int col = vcol0 + t * 16 + l16;
    float bias = bv[col];
#pragma unroll
    for (int j = 0; j < 8; ++j) {
      int n = nbase + rowb + j + 8 * half;
      vg[(b * CCH + col) * NTOK + n] = (__bf16)(accV[t][j] + bias);
    }
  }
}

// ---------------------------------------------------------------------------
// Kernel 2: flash attention + residual. One workgroup = (batch, 64 query rows).
// Wave w owns query row-tile (w&3) and C-column half (w>>2). K/V tiles are
// TDM-DMA'd into double-buffered LDS (prefetch of tile mt+1 overlaps compute
// on tile mt); online softmax; P bounced through per-wave LDS (D->A layout).
// ---------------------------------------------------------------------------
#define KT_STRIDE 40    // 32 r, padded: 80B rows  (64B data + 16B pad)
#define VT_STRIDE 72    // 64 tokens, padded: 144B rows (128B data + 16B pad)
#define PB_STRIDE 72
#define KT_ELEMS (64 * KT_STRIDE)
#define VT_ELEMS (256 * VT_STRIDE)

__global__ __launch_bounds__(256)
void flash_kernel(const float* __restrict__ x,
                  const __bf16* __restrict__ qg,
                  const __bf16* __restrict__ kg,
                  const __bf16* __restrict__ vg,
                  const float* __restrict__ gamma,
                  float* __restrict__ out) {
  __shared__ __align__(16) __bf16 kt[2][KT_ELEMS];           // 2 x  5 KB
  __shared__ __align__(16) __bf16 vt[2][VT_ELEMS];           // 2 x 36 KB
  __shared__ __align__(16) __bf16 pbuf[8 * 16 * PB_STRIDE];  // 18 KB

  const int b      = blockIdx.x >> 6;
  const int nqbase = (blockIdx.x & 63) * 64;
  const int tid    = threadIdx.x;
  const int w      = tid >> 5;
  const int lane   = tid & 31;
  const int l16    = lane & 15;
  const int half   = lane >> 4;
  const int rt     = w & 3;
  const int chalf  = w >> 2;

  // Q A-fragment for this wave's 16 rows (lives in registers all kernel).
  const int qrow = nqbase + rt * 16 + l16;
  const __bf16* qp = &qg[(b * NTOK + qrow) * RDIM];
  const v16bf aq = frag16(qp + half * 16, qp + half * 16 + 8);

  const float scale = 0.17677669529663687f;  // 1/sqrt(32)

  float mrow[8], lrow[8];
  v8f acc[8];
#pragma unroll
  for (int j = 0; j < 8; ++j) { mrow[j] = -1e30f; lrow[j] = 0.f; }
#pragma unroll
  for (int t = 0; t < 8; ++t) acc[t] = {};

  __bf16* pb = &pbuf[w * 16 * PB_STRIDE];

#if USE_TDM
  // Issue the per-wave TDM descriptors for tile mt into LDS buffer `buf`:
  // each wave DMAs its 32-channel slice of V (64 tok x 32 ch, rows padded
  // 128B+16B); wave 0 also DMAs the K tile (32 r x 64 tok, rows 64B+16B).
  auto issue_tile = [&](int mt, int buf) {
    int mbase = mt * 64;
    tdm_load_2d_bf16((uint32_t)(uintptr_t)&vt[buf][0]
                         + (uint32_t)(w * 32 * VT_STRIDE * 2),
                     &vg[(size_t)(b * CCH + w * 32) * NTOK + mbase],
                     /*tile_d0=*/64, /*tile_d1=*/32,
                     /*tensor_d0=*/NTOK, /*tensor_d1=*/CCH, /*stride_d0=*/NTOK,
                     /*pad_interval(32DW)=*/4, /*pad_amount(4DW)=*/3);
    if (w == 0) {
      tdm_load_2d_bf16((uint32_t)(uintptr_t)&kt[buf][0],
                       &kg[(size_t)(b * NTOK + mbase) * RDIM],
                       /*tile_d0=*/32, /*tile_d1=*/64,
                       /*tensor_d0=*/RDIM, /*tensor_d1=*/NTOK,
                       /*stride_d0=*/RDIM,
                       /*pad_interval(16DW)=*/3, /*pad_amount(4DW)=*/3);
    }
  };
#endif

  // Compute one m-tile from LDS buffers ktp/vtp (S -> online softmax -> O).
  auto compute_tile = [&](const __bf16* ktp, const __bf16* vtp) {
    // S = q k^T (K = r = 32 -> one WMMA per 16x16 tile).
    v8f s[4];
    {
      v16bf bk_[4];
#pragma unroll
      for (int mc = 0; mc < 4; ++mc) {
        const __bf16* kp = &ktp[(mc * 16 + l16) * KT_STRIDE + half * 16];
        bk_[mc] = frag16(kp, kp + 8);
      }
#pragma unroll
      for (int mc = 0; mc < 4; ++mc) {
        v8f z = {};
        s[mc] = wmma_bf16(aq, bk_[mc], z);
      }
    }

    // Online softmax. D layout: VGPR j holds rows j (lanes 0-15) / j+8
    // (lanes 16-31); columns across 16 lanes -> reduce with xor 1,2,4,8.
    float nm[8];
#pragma unroll
    for (int j = 0; j < 8; ++j) nm[j] = mrow[j];
#pragma unroll
    for (int mc = 0; mc < 4; ++mc)
#pragma unroll
      for (int j = 0; j < 8; ++j) {
        s[mc][j] *= scale;
        nm[j] = fmaxf(nm[j], s[mc][j]);
      }
#pragma unroll
    for (int mask = 1; mask <= 8; mask <<= 1)
#pragma unroll
      for (int j = 0; j < 8; ++j)
        nm[j] = fmaxf(nm[j], __shfl_xor(nm[j], mask, 32));

    float alpha[8], rs[8];
#pragma unroll
    for (int j = 0; j < 8; ++j) {
      alpha[j] = __expf(mrow[j] - nm[j]);
      mrow[j] = nm[j];
      rs[j] = 0.f;
    }
#pragma unroll
    for (int mc = 0; mc < 4; ++mc)
#pragma unroll
      for (int j = 0; j < 8; ++j) {
        float p = __expf(s[mc][j] - nm[j]);
        s[mc][j] = p;
        rs[j] += p;
      }
#pragma unroll
    for (int mask = 1; mask <= 8; mask <<= 1)
#pragma unroll
      for (int j = 0; j < 8; ++j)
        rs[j] += __shfl_xor(rs[j], mask, 32);
#pragma unroll
    for (int j = 0; j < 8; ++j) lrow[j] = lrow[j] * alpha[j] + rs[j];
#pragma unroll
    for (int t = 0; t < 8; ++t)
#pragma unroll
      for (int j = 0; j < 8; ++j) acc[t][j] *= alpha[j];

    // Bounce P through per-wave LDS: D layout -> A layout (K = 64 m-tokens).
#pragma unroll
    for (int mc = 0; mc < 4; ++mc)
#pragma unroll
      for (int j = 0; j < 8; ++j) {
        int row = j + 8 * half;
        pb[row * PB_STRIDE + mc * 16 + l16] = (__bf16)s[mc][j];
      }
    asm volatile("s_wait_dscnt 0" ::: "memory");  // per-wave RAW on LDS

    const __bf16* pr = &pb[l16 * PB_STRIDE];
    v16bf a0 = frag16(pr + half * 16,      pr + half * 16 + 8);
    v16bf a1 = frag16(pr + 32 + half * 16, pr + 32 + half * 16 + 8);

    // O += P V^T : two WMMAs (K=64) per 16x16 C-tile, groups of 4 tiles.
#pragma unroll
    for (int g2 = 0; g2 < 2; ++g2) {
      v16bf bv0[4], bv1[4];
#pragma unroll
      for (int t = 0; t < 4; ++t) {
        int c = chalf * 128 + (g2 * 4 + t) * 16 + l16;
        const __bf16* vp = &vtp[c * VT_STRIDE];
        bv0[t] = frag16(vp + half * 16,      vp + half * 16 + 8);
        bv1[t] = frag16(vp + 32 + half * 16, vp + 32 + half * 16 + 8);
      }
#pragma unroll
      for (int t = 0; t < 4; ++t) {
        acc[g2 * 4 + t] = wmma_bf16(a0, bv0[t], acc[g2 * 4 + t]);
        acc[g2 * 4 + t] = wmma_bf16(a1, bv1[t], acc[g2 * 4 + t]);
      }
    }
  };

#if USE_TDM
  // Software pipeline: tile mt+1 DMA overlaps tile mt compute.
  issue_tile(0, 0);
  __builtin_amdgcn_s_wait_tensorcnt(0);
  __syncthreads();            // tile 0 resident for all waves
  issue_tile(1, 1);           // prefetch tile 1 into the other buffer
  for (int mt = 0; mt < 64; ++mt) {
    const int cur = mt & 1;
    compute_tile(&kt[cur][0], &vt[cur][0]);
    __builtin_amdgcn_s_wait_tensorcnt(0);  // own slice of tile mt+1 done
    __syncthreads();          // all slices done + buf[cur] fully consumed
    if (mt + 2 < 64) issue_tile(mt + 2, cur);
  }
#else
  // Fallback: manual single-buffered staging with b128 loads/stores.
  for (int mt = 0; mt < 64; ++mt) {
    const int mbase = mt * 64;
    __syncthreads();
    {
      int token = tid >> 2;
      int off   = (tid & 3) * 8;
      *(v8bf*)&kt[0][token * KT_STRIDE + off] =
          *(const v8bf*)&kg[(b * NTOK + mbase + token) * RDIM + off];
    }
#pragma unroll
    for (int i = 0; i < 8; ++i) {
      int chunk = i * 256 + tid;
      int c     = chunk >> 3;
      int off   = (chunk & 7) * 8;
      *(v8bf*)&vt[0][c * VT_STRIDE + off] =
          *(const v8bf*)&vg[(b * CCH + c) * NTOK + mbase + off];
    }
    __syncthreads();
    compute_tile(&kt[0][0], &vt[0][0]);
  }
#endif

  // Epilogue: normalize, residual, store.
  const float g = gamma[0];
#pragma unroll
  for (int t = 0; t < 8; ++t) {
    int c = chalf * 128 + t * 16 + l16;
#pragma unroll
    for (int j = 0; j < 8; ++j) {
      int n = nqbase + rt * 16 + j + 8 * half;
      long idx = (long)(b * CCH + c) * NTOK + n;
      out[idx] = x[idx] + g * (acc[t][j] / lrow[j]);
    }
  }
}

// ---------------------------------------------------------------------------
// Launcher. Workspace layout (needs ~20.2 MB):
//   [0,16K) wq bf16   [16K,32K) wk bf16   [32K,160K) wv bf16
//   [160K,+2M) q bf16 (B,N,32)   [+2M,+4M) k bf16 (B,N,32)
//   [+4M,+20M) v bf16 (B,C,N)
// ---------------------------------------------------------------------------
extern "C" void kernel_launch(void* const* d_in, const int* in_sizes, int n_in,
                              void* d_out, int out_size, void* d_ws, size_t ws_size,
                              hipStream_t stream) {
  const float* x     = (const float*)d_in[0];
  const float* wq    = (const float*)d_in[1];
  const float* bq    = (const float*)d_in[2];
  const float* wk    = (const float*)d_in[3];
  const float* bk    = (const float*)d_in[4];
  const float* wv    = (const float*)d_in[5];
  const float* bv    = (const float*)d_in[6];
  const float* gamma = (const float*)d_in[7];
  float* out = (float*)d_out;

  char* ws = (char*)d_ws;
  __bf16* wq_b = (__bf16*)(ws);
  __bf16* wk_b = (__bf16*)(ws + (16 << 10));
  __bf16* wv_b = (__bf16*)(ws + (32 << 10));
  __bf16* qb   = (__bf16*)(ws + (160 << 10));
  __bf16* kb   = (__bf16*)(ws + (160 << 10) + (2 << 20));
  __bf16* vb   = (__bf16*)(ws + (160 << 10) + (4 << 20));

  convert_weights<<<256, 256, 0, stream>>>(wq, wk, wv, wq_b, wk_b, wv_b);
  proj_kernel<<<BATCH * (NTOK / 64), 256, 0, stream>>>(
      x, wq_b, wk_b, wv_b, bq, bk, bv, qb, kb, vb);
  flash_kernel<<<BATCH * (NTOK / 64), 256, 0, stream>>>(
      x, qb, kb, vb, gamma, out);
}
